// FullyConnectedTensorProduct_87677462380900
// MI455X (gfx1250) — matrix operand order repaired and checked by reading, not compiled, not run on hardware
//
#include <hip/hip_runtime.h>
#include <cstdint>
#include <cmath>
#include <complex>

// ---------------------------------------------------------------------------
// e3nn FullyConnectedTensorProduct (128x0e+128x1o+128x2e) x (1x0e+1x1o+1x2e)
//  -> (128x0e+128x1o+128x2e), B=32768.
// Per output irrep io: out[(b,k),w] = Y_io[(b,k),u'] @ W'_io[u',w], u' = stacked
// (path,u).  GEMMs via v_wmma_f32_16x16x32_bf16 with an error-compensated
// hi/lo bf16 split (3 WMMAs per fp32 product).  Y staged in LDS as two bf16
// planes (hi, lo) so A fragments load directly with ds_load_b128 (no unpack).
// ---------------------------------------------------------------------------

typedef __attribute__((ext_vector_type(16))) __bf16 v16bf;
typedef __attribute__((ext_vector_type(8)))  float  v8f;

// path tables: (i1,i2,io) per reference INSTRUCTIONS order
constexpr int P_I1[11] = {0,0,0,1,1,1,1,2,2,2,2};
constexpr int P_I2[11] = {0,1,2,0,1,1,2,0,1,2,2};
constexpr int P_IO[11] = {0,1,2,1,0,2,1,2,1,0,2};
// flat offsets of each path's w3j tensor (row-major [i][j][k])
constexpr int W3OFF[11] = {0,1,10,35,44,53,98,143,168,213,238};
constexpr int W3TOT = 363;

struct W3JArg { float w[W3TOT]; };

constexpr int NB    = 16;           // batch elements per sub-tile
constexpr int BATCH = 32768;
constexpr int NSUB  = BATCH / NB;   // 2048
constexpr int NUM_WG = 512;
constexpr int DIM1  = 1152;

// ---------------- device helpers ----------------
__device__ __forceinline__ uint32_t f2bf(float f) {
  uint32_t u = __float_as_uint(f);
  return (u + 0x7FFFu + ((u >> 16) & 1u)) >> 16;   // RNE to bf16 bits
}
__device__ __forceinline__ float bf2f(uint32_t b) { return __uint_as_float(b << 16); }

__device__ __forceinline__ constexpr int s1off(int i1) { return i1 == 0 ? 0 : (i1 == 1 ? 128 : 512); }
__device__ __forceinline__ constexpr int s2off(int i2) { return i2 == 0 ? 0 : (i2 == 1 ? 1 : 4); }

// XOR-swizzled LDS dword index inside one bf16 plane.
// Plane row = KTOT bf16 = KTOT/2 dwords (KTOT/2 % 64 == 0), 4-dword granules
// xor'ed by (m&15): 16 lanes of a b128 read tile all 64 banks.
template<int KTOT>
__device__ __forceinline__ int yidx16(int m, int dc) {
  return m * (KTOT / 2) + ((((dc >> 2) ^ (m & 15))) << 2) + (dc & 3);
}

union Frag { v16bf v; uint4 q[2]; };

// Build Y rows for one path; each thread handles u pairs (u0,u0+1) so two
// bf16 pack into one dword per plane.
template<int P, int I1, int I2, int LO, int PL, int KTOT>
__device__ __forceinline__ void path_y(const float* __restrict__ x1row,
                                       const float* x2v, const W3JArg& w3,
                                       uint32_t* HiP, uint32_t* LoP,
                                       int bl, int l16) {
  constexpr int N1 = 2 * I1 + 1, N2 = 2 * I2 + 1, NO = 2 * LO + 1;
  float M[N1][NO];
#pragma unroll
  for (int i = 0; i < N1; i++) {
#pragma unroll
    for (int k = 0; k < NO; k++) {
      float s = 0.f;
#pragma unroll
      for (int j = 0; j < N2; j++)
        s += w3.w[W3OFF[P] + (i * N2 + j) * NO + k] * x2v[s2off(I2) + j];
      M[i][k] = s;
    }
  }
#pragma unroll
  for (int r = 0; r < 4; r++) {
    const int u0 = 2 * l16 + 32 * r;
    const float* xp = x1row + s1off(I1) + u0 * N1;
    float xv[2 * N1];
#pragma unroll
    for (int i = 0; i < 2 * N1; i++) xv[i] = xp[i];
#pragma unroll
    for (int k = 0; k < NO; k++) {
      float y0 = 0.f, y1 = 0.f;
#pragma unroll
      for (int i = 0; i < N1; i++) {
        y0 += xv[i] * M[i][k];
        y1 += xv[N1 + i] * M[i][k];
      }
      const uint32_t h0 = f2bf(y0), h1 = f2bf(y1);
      const uint32_t l0 = f2bf(y0 - bf2f(h0)), l1 = f2bf(y1 - bf2f(h1));
      const int idx = yidx16<KTOT>(bl * NO + k, (PL * 128 + u0) >> 1);
      HiP[idx] = h0 | (h1 << 16);
      LoP[idx] = l0 | (l1 << 16);
    }
  }
}

template<int IO>
__device__ void stage(const float* __restrict__ x1, const float* __restrict__ x2,
                      float* __restrict__ out, const uint16_t* __restrict__ wpack,
                      uint32_t* Ylds, const W3JArg& w3) {
  constexpr int OD    = 2 * IO + 1;
  constexpr int NPATH = (IO == 0) ? 3 : 4;
  constexpr int KTOT  = NPATH * 128;          // 384 / 512
  constexpr int KT    = KTOT / 32;            // 12 / 16
  constexpr int MT    = OD;                   // NB*OD/16 = OD (NB==16)
  constexpr int FBASE = (IO == 0) ? 0 : ((IO == 1) ? 96 : 224);
  constexpr int OBASE = (IO == 0) ? 0 : ((IO == 1) ? 128 : 512);

  uint32_t* HiP = Ylds;
  uint32_t* LoP = Ylds + (MT * 16) * (KTOT / 2);

  const int tid  = threadIdx.x;
  const int lane = tid & 31;
  const int nt   = tid >> 5;                  // wave -> N tile (8 waves, N=128)

  // B fragments (weights), register-resident for the whole stage.
  Frag bhi[KT], blo[KT];
#pragma unroll
  for (int kt = 0; kt < KT; kt++) {
    const uint16_t* ph = wpack + (size_t)(FBASE + kt * 8 + nt) * 1024u + (size_t)lane * 16u;
    bhi[kt].q[0] = ((const uint4*)ph)[0];
    bhi[kt].q[1] = ((const uint4*)ph)[1];
    blo[kt].q[0] = ((const uint4*)(ph + 512))[0];
    blo[kt].q[1] = ((const uint4*)(ph + 512))[1];
  }

  const int l15     = lane & 15;
  const int laneoff = (lane >= 16) ? 16 : 0;  // A layout K sub-offset (bf16 cols)

  for (int st = blockIdx.x; st < NSUB; st += gridDim.x) {
    const int b0 = st * NB;
    { // ---- build Y in LDS (all 256 threads) ----
      const int bl  = tid >> 4;
      const int l16 = tid & 15;
      const int b   = b0 + bl;
      float x2v[9];
#pragma unroll
      for (int j = 0; j < 9; j++) x2v[j] = x2[(size_t)b * 9 + j];
      const float* x1row = x1 + (size_t)b * DIM1;
      if constexpr (IO == 0) {
        path_y<0, 0, 0, 0, 0, KTOT>(x1row, x2v, w3, HiP, LoP, bl, l16);
        path_y<4, 1, 1, 0, 1, KTOT>(x1row, x2v, w3, HiP, LoP, bl, l16);
        path_y<9, 2, 2, 0, 2, KTOT>(x1row, x2v, w3, HiP, LoP, bl, l16);
      } else if constexpr (IO == 1) {
        path_y<1, 0, 1, 1, 0, KTOT>(x1row, x2v, w3, HiP, LoP, bl, l16);
        path_y<3, 1, 0, 1, 1, KTOT>(x1row, x2v, w3, HiP, LoP, bl, l16);
        path_y<6, 1, 2, 1, 2, KTOT>(x1row, x2v, w3, HiP, LoP, bl, l16);
        path_y<8, 2, 1, 1, 3, KTOT>(x1row, x2v, w3, HiP, LoP, bl, l16);
      } else {
        path_y<2, 0, 2, 2, 0, KTOT>(x1row, x2v, w3, HiP, LoP, bl, l16);
        path_y<5, 1, 1, 2, 1, KTOT>(x1row, x2v, w3, HiP, LoP, bl, l16);
        path_y<7, 2, 0, 2, 2, KTOT>(x1row, x2v, w3, HiP, LoP, bl, l16);
        path_y<10, 2, 2, 2, 3, KTOT>(x1row, x2v, w3, HiP, LoP, bl, l16);
      }
    }
    __syncthreads();

    // ---- GEMM: Y[(b,k), u'] @ W'[u', w] via 3-term bf16 split ----
    const int w = nt * 16 + l15;
    for (int mt = 0; mt < MT; ++mt) {
      const int m = mt * 16 + l15;            // A row for this lane
      v8f acc = {};
#pragma unroll
      for (int kt = 0; kt < KT; kt++) {
        // A element j <-> K = j + laneoff : two 8-bf16 runs.
        const int dc0 = kt * 16 + (laneoff >> 1);       // dword col of run 0
        const int i0 = yidx16<KTOT>(m, dc0);
        const int i1 = yidx16<KTOT>(m, dc0 + 4);        // run 1 (+8 bf16)
        Frag ah, al;
        ah.q[0] = *(const uint4*)(HiP + i0);
        ah.q[1] = *(const uint4*)(HiP + i1);
        al.q[0] = *(const uint4*)(LoP + i0);
        al.q[1] = *(const uint4*)(LoP + i1);
        acc = __builtin_amdgcn_wmma_f32_16x16x32_bf16(false, ah.v, false, bhi[kt].v, (short)0, acc, false, false);
        acc = __builtin_amdgcn_wmma_f32_16x16x32_bf16(false, ah.v, false, blo[kt].v, (short)0, acc, false, false);
        acc = __builtin_amdgcn_wmma_f32_16x16x32_bf16(false, al.v, false, bhi[kt].v, (short)0, acc, false, false);
      }
#pragma unroll
      for (int r = 0; r < 8; r++) {
        const int mrow = mt * 16 + ((lane < 16) ? r : (8 + r));
        const int blc = mrow / OD;
        const int k   = mrow - blc * OD;
        out[(size_t)(b0 + blc) * DIM1 + OBASE + (size_t)w * OD + k] = acc[r];
      }
    }
    __syncthreads();
  }
}

__global__ __launch_bounds__(256)
void tp_main(const float* __restrict__ x1, const float* __restrict__ x2,
             float* __restrict__ out, const uint16_t* __restrict__ wpack,
             W3JArg w3) {
  extern __shared__ uint32_t Ylds[];
  stage<0>(x1, x2, out, wpack, Ylds, w3);
  stage<1>(x1, x2, out, wpack, Ylds, w3);
  stage<2>(x1, x2, out, wpack, Ylds, w3);
}

// Pack sqrt(alpha)*ws into WMMA-B fragment layout, bf16 hi/lo.
// frag f in [0,352): io blocks {0..95, 96..223, 224..351}; within: kt*8+nt.
// Per frag: 512 hi bf16 then 512 lo bf16; lane L owns 16 values (32B each half).
// B layout: lane<16 -> N=lane, Klocal=j ; lane>=16 -> N=lane-16, Klocal=16+j.
__global__ void tp_prep(const float* __restrict__ ws, uint16_t* __restrict__ wpack) {
  const int f = blockIdx.x;
  const int lane = threadIdx.x;
  int io, rem;
  if (f < 96)       { io = 0; rem = f; }
  else if (f < 224) { io = 1; rem = f - 96; }
  else              { io = 2; rem = f - 224; }
  const int kt = rem >> 3, nt = rem & 7;
  const int n  = nt * 16 + (lane & 15);
  const int kb = kt * 32 + ((lane >= 16) ? 16 : 0);
  const float sa = sqrtf(io == 0 ? (1.f / 384.f) : (io == 1 ? (3.f / 512.f) : (5.f / 512.f)));
  const int paths[3][4] = {{0, 4, 9, 0}, {1, 3, 6, 8}, {2, 5, 7, 10}};
  union { uint16_t s[16]; uint4 q[2]; } H, L;
#pragma unroll
  for (int j = 0; j < 16; j++) {
    const int up = kb + j;          // stacked u' = pl*128 + u
    const int pl = up >> 7, u = up & 127;
    const int p = paths[io][pl];
    const float v = sa * ws[((size_t)p * 128 + u) * 128 + n];
    const uint32_t hb = f2bf(v);
    const uint32_t lb = f2bf(v - bf2f(hb));
    H.s[j] = (uint16_t)hb;
    L.s[j] = (uint16_t)lb;
  }
  uint16_t* dst = wpack + (size_t)f * 1024u + (size_t)lane * 16u;
  ((uint4*)dst)[0] = H.q[0];
  ((uint4*)dst)[1] = H.q[1];
  ((uint4*)(dst + 512))[0] = L.q[0];
  ((uint4*)(dst + 512))[1] = L.q[1];
}

// ---------------- host: real Wigner-3j (port of reference) ----------------
static double fct(int n) { double r = 1.0; for (int i = 2; i <= n; i++) r *= (double)i; return r; }

static void su2_cg(int j1, int j2, int j3, double C[5][5][5]) {
  for (int a = 0; a < 5; a++) for (int b = 0; b < 5; b++) for (int c = 0; c < 5; c++) C[a][b][c] = 0.0;
  for (int m1 = -j1; m1 <= j1; m1++)
    for (int m2 = -j2; m2 <= j2; m2++) {
      const int m3 = m1 + m2;
      if (std::abs(m3) > j3) continue;
      double pref = std::sqrt((2 * j3 + 1) * fct(j3 + j1 - j2) * fct(j3 - j1 + j2) * fct(j1 + j2 - j3) / fct(j1 + j2 + j3 + 1));
      pref *= std::sqrt(fct(j3 + m3) * fct(j3 - m3) * fct(j1 - m1) * fct(j1 + m1) * fct(j2 - m2) * fct(j2 + m2));
      double s = 0.0;
      for (int k = 0; k <= j1 + j2 - j3; k++) {
        const int a = j1 - m1 - k, b = j2 + m2 - k, c = j3 - j2 + m1 + k, d = j3 - j1 - m2 + k;
        if (a < 0 || b < 0 || c < 0 || d < 0) continue;
        s += ((k & 1) ? -1.0 : 1.0) / (fct(k) * fct(j1 + j2 - j3 - k) * fct(a) * fct(b) * fct(c) * fct(d));
      }
      C[j1 + m1][j2 + m2][j3 + m3] = pref * s;
    }
}

typedef std::complex<double> cdbl;
static void real_basis(int l, cdbl q[5][5]) {
  for (int a = 0; a < 5; a++) for (int b = 0; b < 5; b++) q[a][b] = cdbl(0.0, 0.0);
  const double is2 = 1.0 / std::sqrt(2.0);
  for (int m = -l; m < 0; m++) {
    q[l + m][l - m] = cdbl(is2, 0.0);
    q[l + m][l + m] = cdbl(0.0, -is2);
  }
  q[l][l] = cdbl(1.0, 0.0);
  for (int m = 1; m <= l; m++) {
    const double sg = (m & 1) ? -1.0 : 1.0;
    q[l + m][l + m] = cdbl(sg * is2, 0.0);
    q[l + m][l - m] = cdbl(0.0, sg * is2);
  }
  cdbl f(1.0, 0.0); const cdbl mi(0.0, -1.0);
  for (int t = 0; t < l; t++) f *= mi;     // (-1j)^l
  for (int a = 0; a < 5; a++) for (int b = 0; b < 5; b++) q[a][b] *= f;
}

static void w3j_fill(int l1, int l2, int l3, float* outp) {
  double C[5][5][5]; su2_cg(l1, l2, l3, C);
  cdbl Q1[5][5], Q2[5][5], Q3[5][5];
  real_basis(l1, Q1); real_basis(l2, Q2); real_basis(l3, Q3);
  const int d1 = 2 * l1 + 1, d2 = 2 * l2 + 1, d3 = 2 * l3 + 1;
  double nrm = 0.0; double R[5][5][5];
  for (int j = 0; j < d1; j++) for (int lx = 0; lx < d2; lx++) for (int m = 0; m < d3; m++) {
    cdbl s(0.0, 0.0);
    for (int i = 0; i < d1; i++) for (int k = 0; k < d2; k++) for (int n = 0; n < d3; n++)
      s += Q1[i][j] * Q2[k][lx] * std::conj(Q3[n][m]) * C[i][k][n];
    R[j][lx][m] = s.real(); nrm += s.real() * s.real();
  }
  nrm = std::sqrt(nrm);
  for (int j = 0; j < d1; j++) for (int lx = 0; lx < d2; lx++) for (int m = 0; m < d3; m++)
    outp[(j * d2 + lx) * d3 + m] = (float)(R[j][lx][m] / nrm);
}

// ---------------------------------------------------------------------------
extern "C" void kernel_launch(void* const* d_in, const int* in_sizes, int n_in,
                              void* d_out, int out_size, void* d_ws, size_t ws_size,
                              hipStream_t stream) {
  (void)in_sizes; (void)n_in; (void)out_size; (void)ws_size;
  const float* x1 = (const float*)d_in[0];
  const float* x2 = (const float*)d_in[1];
  const float* ws = (const float*)d_in[2];
  float* out = (float*)d_out;
  uint16_t* wpack = (uint16_t*)d_ws;

  W3JArg w3;
  for (int p = 0; p < 11; p++)
    w3j_fill(P_I1[p], P_I2[p], P_IO[p], w3.w + W3OFF[p]);

  tp_prep<<<352, 32, 0, stream>>>(ws, wpack);
  // dynamic LDS: hi+lo planes, max stage (io=2): 80 rows * 512 bf16 * 2 planes
  // * 2B = 160 KB
  tp_main<<<NUM_WG, 256, 80 * 512 * 4, stream>>>(x1, x2, out, wpack, w3);
}
